// SwitchMixtureOfExperts_8478265442765
// MI455X (gfx1250) — matrix operand order repaired and checked by reading, not compiled, not run on hardware
//
#include <hip/hip_runtime.h>
#include <hip/hip_bf16.h>
#include <stdint.h>

#define N_TOK (8 * 4096)
#define DIM   512
#define HID   2048
#define OUTD  512
#define NEXP  8
#define MTILE 32          // tokens per block (2 WMMA M-subtiles, B-fragment reuse x2)

// Workspace layout (bytes)
#define OFF_COUNTS 0
#define OFF_LIST   1024
#define OFF_XB     (OFF_LIST + (size_t)NEXP * N_TOK * 4)
#define OFF_W1B    (OFF_XB + (size_t)N_TOK * DIM * 2)
#define OFF_W2B    (OFF_W1B + (size_t)NEXP * DIM * HID * 2)

typedef __attribute__((ext_vector_type(16))) __bf16 v16bf;
typedef __attribute__((ext_vector_type(8)))  float  v8f;

union V16 {
  v16bf v;
  uint4 q[2];
  unsigned short u[16];
};

__device__ __forceinline__ unsigned short f2bf(float f) {
  unsigned int u = __builtin_bit_cast(unsigned int, f);
  u += 0x7FFFu + ((u >> 16) & 1u);   // round-to-nearest-even
  return (unsigned short)(u >> 16);
}

__global__ void zero_counts_kernel(int* counts) {
  if (threadIdx.x < NEXP) counts[threadIdx.x] = 0;
}

__global__ void f32_to_bf16_kernel(const float* __restrict__ in,
                                   unsigned short* __restrict__ out, int n) {
  int i = blockIdx.x * blockDim.x + threadIdx.x;
  int stride = gridDim.x * blockDim.x;
  for (; i < n; i += stride) out[i] = f2bf(in[i]);
}

// One thread per token: logits = x_row @ Wr + br ; argmax ; append to expert list.
__global__ void router_kernel(const float* __restrict__ x,
                              const float* __restrict__ Wr,
                              const float* __restrict__ br,
                              int* __restrict__ counts,
                              int* __restrict__ list) {
  __shared__ float sW[DIM * NEXP];
  for (int i = threadIdx.x; i < DIM * NEXP; i += blockDim.x) sW[i] = Wr[i];
  __syncthreads();

  int t = blockIdx.x * blockDim.x + threadIdx.x;
  if (t >= N_TOK) return;
  const float* row = x + (size_t)t * DIM;

  float acc[NEXP];
#pragma unroll
  for (int e = 0; e < NEXP; ++e) acc[e] = br[e];
  for (int j = 0; j < DIM; ++j) {
    float xv = row[j];
#pragma unroll
    for (int e = 0; e < NEXP; ++e) acc[e] += xv * sW[j * NEXP + e];
  }
  int best = 0; float bv = acc[0];
#pragma unroll
  for (int e = 1; e < NEXP; ++e)
    if (acc[e] > bv) { bv = acc[e]; best = e; }

  int slot = atomicAdd(&counts[best], 1);
  list[best * N_TOK + slot] = t;
  // TOP_K==1 -> softmax over one logit -> gate == 1.0 exactly.
}

// A fragment: 16x32 bf16 tile from row-major LDS (ISA 16-bit A layout).
// lane 0-15: M=lane, elems 0..7 -> K=kb+0..7, elems 8..15 -> K=kb+16..23 (kb=0)
// lane 16-31: same with kb=8.
__device__ __forceinline__ v16bf load_a(const unsigned short* lds, int stride,
                                        int k0, int lane) {
  int m  = lane & 15;
  int kb = (lane >> 4) << 3;
  const unsigned short* p = lds + m * stride + k0 + kb;
  V16 a;
  a.q[0] = *(const uint4*)(p);
  a.q[1] = *(const uint4*)(p + 16);
  return a.v;
}

// B fragment: 32x16 bf16 tile from row-major (K-major) global weights.
// lane%16 = N; elems 0..7 -> K=kb+0..7, elems 8..15 -> K=kb+16..23, kb=8*(lane/16).
__device__ __forceinline__ v16bf load_b(const unsigned short* g, int ldn,
                                        int k0, int n0, int lane) {
  const unsigned short* p = g + (size_t)(k0 + ((lane >> 4) << 3)) * ldn + n0 + (lane & 15);
  V16 b;
#pragma unroll
  for (int s = 0; s < 16; ++s)
    b.u[s] = p[(size_t)((s & 7) + ((s >> 3) << 4)) * ldn];
  return b.v;
}

// One block (8 wave32) per (expert, 32-token tile): out = relu(x@W1+b1)@W2+b2
__launch_bounds__(256)
__global__ void moe_ffn_kernel(const unsigned short* __restrict__ xb,
                               const unsigned short* __restrict__ w1b,
                               const unsigned short* __restrict__ w2b,
                               const float* __restrict__ b1,
                               const float* __restrict__ b2,
                               const int* __restrict__ counts,
                               const int* __restrict__ list,
                               float* __restrict__ out) {
  extern __shared__ unsigned char smem[];
  unsigned short* sX   = (unsigned short*)smem;                            // 32*512 bf16
  unsigned short* sH   = (unsigned short*)(smem + MTILE * DIM * 2);        // 32*2048 bf16
  int*            sTok = (int*)(smem + MTILE * DIM * 2 + MTILE * HID * 2); // 32 ints

  const int e    = blockIdx.y;
  const int cnt  = counts[e];
  const int base = blockIdx.x * MTILE;
  if (base >= cnt) return;                 // uniform early-exit for empty tiles
  int validCnt = cnt - base; if (validCnt > MTILE) validCnt = MTILE;

  if (threadIdx.x < MTILE) {
    int slot = base + (int)threadIdx.x;
    if (slot >= cnt) slot = cnt - 1;       // clamp pad rows to a valid token
    sTok[threadIdx.x] = list[e * N_TOK + slot];
  }
  __syncthreads();

  // Gather 32 token rows of bf16 x into LDS: 256 threads * 128B each.
  {
    int r = threadIdx.x >> 3;              // row 0..31
    int c = (threadIdx.x & 7) * 64;        // 64 bf16 per thread
    const unsigned short* src = xb + (size_t)sTok[r] * DIM + c;
    unsigned short* dst = sX + r * DIM + c;
#pragma unroll
    for (int q = 0; q < 8; ++q)
      *(uint4*)(dst + q * 8) = *(const uint4*)(src + q * 8);
  }
  __syncthreads();

  const int lane = threadIdx.x & 31;
  const int wave = threadIdx.x >> 5;
  const int n    = lane & 15;              // D/C column
  const int mb   = (lane >> 4) << 3;       // D/C row base (0 or 8)
  const unsigned short* W1e = w1b + (size_t)e * DIM * HID;
  const unsigned short* W2e = w2b + (size_t)e * HID * OUTD;

  // GEMM1: sH = relu(sX[32x512] @ W1e[512x2048] + b1[e])
  // 128 N-tiles / 8 waves; each B fragment reused for both 16-row M-subtiles.
  for (int nt = wave; nt < HID / 16; nt += 8) {
    v8f acc0 = {}, acc1 = {};
    for (int k0 = 0; k0 < DIM; k0 += 32) {
      v16bf b  = load_b(W1e, HID, k0, nt * 16, lane);
      v16bf a0 = load_a(sX,             DIM, k0, lane);
      v16bf a1 = load_a(sX + 16 * DIM,  DIM, k0, lane);
      acc0 = __builtin_amdgcn_wmma_f32_16x16x32_bf16(false, a0, false, b,
                                                     (short)0, acc0, false, false);
      acc1 = __builtin_amdgcn_wmma_f32_16x16x32_bf16(false, a1, false, b,
                                                     (short)0, acc1, false, false);
    }
    float bias = b1[e * HID + nt * 16 + n];
#pragma unroll
    for (int r = 0; r < 8; ++r) {
      float v0 = acc0[r] + bias;
      float v1 = acc1[r] + bias;
      v0 = v0 > 0.0f ? v0 : 0.0f;
      v1 = v1 > 0.0f ? v1 : 0.0f;
      sH[(mb + r) * HID + nt * 16 + n]        = f2bf(v0);
      sH[(16 + mb + r) * HID + nt * 16 + n]   = f2bf(v1);
    }
  }
  __syncthreads();

  // GEMM2: out = sH[32x2048] @ W2e[2048x512] + b2[e]   (32 N-tiles / 8 waves)
  for (int nt = wave; nt < OUTD / 16; nt += 8) {
    v8f acc0 = {}, acc1 = {};
    for (int k0 = 0; k0 < HID; k0 += 32) {
      v16bf b  = load_b(W2e, OUTD, k0, nt * 16, lane);
      v16bf a0 = load_a(sH,            HID, k0, lane);
      v16bf a1 = load_a(sH + 16 * HID, HID, k0, lane);
      acc0 = __builtin_amdgcn_wmma_f32_16x16x32_bf16(false, a0, false, b,
                                                     (short)0, acc0, false, false);
      acc1 = __builtin_amdgcn_wmma_f32_16x16x32_bf16(false, a1, false, b,
                                                     (short)0, acc1, false, false);
    }
    float bias = b2[e * OUTD + nt * 16 + n];
#pragma unroll
    for (int r = 0; r < 8; ++r) {
      int m0 = mb + r;
      int m1 = 16 + mb + r;
      if (m0 < validCnt)
        out[(size_t)sTok[m0] * OUTD + nt * 16 + n] = acc0[r] + bias;  // gate == 1.0
      if (m1 < validCnt)
        out[(size_t)sTok[m1] * OUTD + nt * 16 + n] = acc1[r] + bias;
    }
  }
}

extern "C" void kernel_launch(void* const* d_in, const int* in_sizes, int n_in,
                              void* d_out, int out_size, void* d_ws, size_t ws_size,
                              hipStream_t stream) {
  (void)in_sizes; (void)n_in; (void)out_size; (void)ws_size;
  const float* x  = (const float*)d_in[0];
  const float* Wr = (const float*)d_in[1];
  const float* br = (const float*)d_in[2];
  const float* W1 = (const float*)d_in[3];
  const float* b1 = (const float*)d_in[4];
  const float* W2 = (const float*)d_in[5];
  const float* b2 = (const float*)d_in[6];
  float* out = (float*)d_out;

  char* ws = (char*)d_ws;
  int* counts         = (int*)(ws + OFF_COUNTS);
  int* list           = (int*)(ws + OFF_LIST);
  unsigned short* xb  = (unsigned short*)(ws + OFF_XB);
  unsigned short* w1b = (unsigned short*)(ws + OFF_W1B);
  unsigned short* w2b = (unsigned short*)(ws + OFF_W2B);

  zero_counts_kernel<<<1, 32, 0, stream>>>(counts);
  f32_to_bf16_kernel<<<2048, 256, 0, stream>>>(x,  xb,  N_TOK * DIM);
  f32_to_bf16_kernel<<<2048, 256, 0, stream>>>(W1, w1b, NEXP * DIM * HID);
  f32_to_bf16_kernel<<<2048, 256, 0, stream>>>(W2, w2b, NEXP * HID * OUTD);
  router_kernel<<<N_TOK / 256, 256, 0, stream>>>(x, Wr, br, counts, list);

  dim3 grid(N_TOK / MTILE, NEXP);
  size_t smem = (size_t)MTILE * DIM * 2 + (size_t)MTILE * HID * 2 + 128; // ~160 KB
  moe_ffn_kernel<<<grid, 256, smem, stream>>>(xb, w1b, w2b, b1, b2,
                                              counts, list, out);
}
